// AKConvModule_66632122630858
// MI455X (gfx1250) — matrix operand order repaired and checked by reading, not compiled, not run on hardware
//
#include <hip/hip_runtime.h>
#include <hip/hip_bf16.h>

// ---------------------------------------------------------------------------
// AKConv forward, fused for MI455X (gfx1250, wave32, WMMA).
//   B=4, C1=128, C2=256, H=W=128, N=9, K = C1*N = 1152.
//   GEMM: M=C2=256, K=1152, Nspatial=65536 -> bf16 WMMA 16x16x32, f32 accum.
// Block = 256 threads (8 waves) owns 64 spatial positions (4 WMMA N-tiles),
// so each L2 weight fragment feeds 4 WMMAs (4x less L2 traffic than 16-pos).
// ---------------------------------------------------------------------------

#define Bn    4
#define C1n   128
#define C2n   256
#define Hn    128
#define Wn    128
#define Nn    9
#define Kdim  (C1n * Nn)          // 1152
#define HWn   (Hn * Wn)           // 16384
#define POS   64                  // spatial positions per block (4 N-tiles)
#define BROW  1176                // LDS row stride (ushorts): 2352B = 588 dw,
                                  // 12*l mod 64 hits distinct 4-dw slots ->
                                  // conflict-free ds_load_b128 across 16 lanes
#define OFF_ELEMS (Bn * 2 * Nn * Hn * Wn)   // 1,179,648 floats
#define WMAIN_ELEMS (C2n * Kdim)            // 294,912

typedef __attribute__((ext_vector_type(8)))  unsigned short v8us;
typedef __attribute__((ext_vector_type(16))) unsigned short v16us;
typedef __attribute__((ext_vector_type(16))) __bf16         v16bf;
typedef __attribute__((ext_vector_type(8)))  float          v8f;

__device__ __forceinline__ unsigned short f32_to_bf16_rne(float f) {
    unsigned int u = __builtin_bit_cast(unsigned int, f);
    unsigned int r = u + 0x7FFFu + ((u >> 16) & 1u);
    return (unsigned short)(r >> 16);
}

// ---------------------------------------------------------------------------
// K1: convert w_main (C2,C1,N) fp32 -> bf16, flat layout [o][c*9+n] == [o][k].
// ---------------------------------------------------------------------------
__global__ void akconv_convert_wmain(const float* __restrict__ wm,
                                     unsigned short* __restrict__ wbf, int n) {
    int i = blockIdx.x * 256 + threadIdx.x;
    if (i < n) wbf[i] = f32_to_bf16_rne(wm[i]);
}

// ---------------------------------------------------------------------------
// K2: offset-prediction conv: 3x3, pad 1, C1 -> 18 channels.
// One block per (b,h); 128 threads across w; all 18 out channels in registers.
// Weight reads are wave-uniform -> scalar loads through the constant path.
// ---------------------------------------------------------------------------
__global__ void akconv_offset_conv(const float* __restrict__ x,
                                   const float* __restrict__ w_off,
                                   const float* __restrict__ b_off,
                                   float* __restrict__ off) {
    const int bh = blockIdx.x;            // 0 .. B*H-1
    const int b  = bh >> 7;
    const int h  = bh & (Hn - 1);
    const int w  = threadIdx.x;           // 0..127

    float acc[2 * Nn];
#pragma unroll
    for (int ch = 0; ch < 2 * Nn; ++ch) acc[ch] = b_off[ch];

    const float* xb = x + (size_t)b * C1n * HWn;
    for (int c = 0; c < C1n; ++c) {
        const float* xc = xb + (size_t)c * HWn;
        float xv[3][3];
#pragma unroll
        for (int kh = 0; kh < 3; ++kh) {
            const int hh = h + kh - 1;
            const bool hok = (hh >= 0) && (hh < Hn);
#pragma unroll
            for (int kw = 0; kw < 3; ++kw) {
                const int ww = w + kw - 1;
                const bool ok = hok && (ww >= 0) && (ww < Wn);
                xv[kh][kw] = ok ? xc[hh * Wn + ww] : 0.0f;
            }
        }
#pragma unroll
        for (int ch = 0; ch < 2 * Nn; ++ch) {
            const float* wp = w_off + ((size_t)(ch * C1n + c)) * 9;
            acc[ch] += xv[0][0] * wp[0] + xv[0][1] * wp[1] + xv[0][2] * wp[2]
                     + xv[1][0] * wp[3] + xv[1][1] * wp[4] + xv[1][2] * wp[5]
                     + xv[2][0] * wp[6] + xv[2][1] * wp[7] + xv[2][2] * wp[8];
        }
    }
#pragma unroll
    for (int ch = 0; ch < 2 * Nn; ++ch)
        off[(((size_t)b * 2 * Nn + ch) * Hn + h) * Wn + w] = acc[ch];
}

// ---------------------------------------------------------------------------
// K3: fused bilinear-sample -> bf16 WMMA GEMM -> BN -> SiLU.
// Block = 256 threads (8 waves), owns 64 consecutive spatial positions
// (always same b and same h, since 64 | W). Each wave computes 32 output
// channels x 64 positions: 2 M-tiles x 4 N-tiles over K=1152 (36 x 32).
// ---------------------------------------------------------------------------
__global__ void __launch_bounds__(256)
akconv_fused_gemm(const float* __restrict__ x,
                  const float* __restrict__ off,
                  const unsigned short* __restrict__ wbf,
                  const float* __restrict__ gamma,
                  const float* __restrict__ beta,
                  const float* __restrict__ mean,
                  const float* __restrict__ var,
                  float* __restrict__ out) {
    const int t  = threadIdx.x;
    const int p0 = blockIdx.x * POS;          // first spatial position
    const int b  = p0 >> 14;                  // / (H*W)
    const int h  = (p0 & (HWn - 1)) >> 7;
    const int w0 = p0 & (Wn - 1);             // 0 or 64

    // bilinear metadata: 9 sample points x 64 positions
    __shared__ int   s_idx[Nn * POS];         // packed ix0|iy0<<8|ix1<<16|iy1<<24
    __shared__ float s_glt[Nn * POS], s_grb[Nn * POS];
    __shared__ float s_glb[Nn * POS], s_grt[Nn * POS];
    __shared__ __align__(16) unsigned short Bsh[POS * BROW];  // B panel: [pos][k]

    // ---- Stage 1: bilinear indices + weights for 64 positions x 9 points ----
    for (int it = t; it < Nn * POS; it += 256) {
        const int j = it & (POS - 1);  // position in tile
        const int n = it / POS;        // sample point 0..8
        const int w = w0 + j;
        const float offx = off[(((size_t)b * 18 + n) * Hn + h) * Wn + w];
        const float offy = off[(((size_t)b * 18 + 9 + n) * Hn + h) * Wn + w];
        const float pnx = (float)(n / 3);
        const float pny = (float)(n % 3);
        float px = fminf(fmaxf(offx + pnx + (float)h, 0.0f), (float)(Hn - 1));
        float py = fminf(fmaxf(offy + pny + (float)w, 0.0f), (float)(Wn - 1));
        float x0f = fminf(fmaxf(floorf(px), 0.0f), (float)(Hn - 1));
        float y0f = fminf(fmaxf(floorf(py), 0.0f), (float)(Wn - 1));
        float x1f = fminf(x0f + 1.0f, (float)(Hn - 1));
        float y1f = fminf(y0f + 1.0f, (float)(Wn - 1));
        s_glt[it] = (1.0f + (x0f - px)) * (1.0f + (y0f - py));
        s_grb[it] = (1.0f - (x1f - px)) * (1.0f - (y1f - py));
        s_glb[it] = (1.0f + (x0f - px)) * (1.0f - (y1f - py));
        s_grt[it] = (1.0f - (x1f - px)) * (1.0f + (y0f - py));
        s_idx[it] = (int)x0f | ((int)y0f << 8) | ((int)x1f << 16) | ((int)y1f << 24);
    }
    __syncthreads();

    // ---- Stage 2: gather + bilinear combine, write bf16 B panel to LDS ----
    // Bsh[pos][k] with k = c*9+n matching w_main's (c,n) flattening.
    const float* xb = x + (size_t)b * C1n * HWn;
    for (int idx = t; idx < C1n * Nn * POS; idx += 256) {
        const int j = idx & (POS - 1);
        const int q = idx >> 6;        // 0..1151
        const int n = q % 9;
        const int c = q / 9;
        const int s = n * POS + j;
        const float* xc = xb + (size_t)c * HWn;
        const int p   = s_idx[s];
        const int ix0 = p & 255,         iy0 = (p >> 8) & 255;
        const int ix1 = (p >> 16) & 255, iy1 = (p >> 24) & 255;
        const float v = s_glt[s] * xc[(ix0 << 7) + iy0]
                      + s_grb[s] * xc[(ix1 << 7) + iy1]
                      + s_glb[s] * xc[(ix0 << 7) + iy1]
                      + s_grt[s] * xc[(ix1 << 7) + iy0];
        Bsh[j * BROW + c * 9 + n] = f32_to_bf16_rne(v);
    }
    __syncthreads();

    // ---- Stage 3: WMMA GEMM over K=1152 (36 x 32), 2 M-tiles x 4 N-tiles ----
    const int lane  = t & 31;
    const int wave  = t >> 5;            // 0..7
    const int m0    = wave * 32;         // 32 output channels per wave
    const int row   = lane & 15;         // A row (M) / B column within tile
    const int khalf = (lane >> 4) * 8;   // A: K offset khalf*2 = 0 or 16, contiguous k = 16h+s

    const unsigned short* wrow0 = wbf + (size_t)(m0 + row) * Kdim;
    const unsigned short* wrow1 = wrow0 + 16 * Kdim;
    const unsigned short* bbase = Bsh + row * BROW + khalf * 2; // B: K off 0/16

    v8f acc[8];                          // [m-tile][n-tile] = acc[m*4+nt]
#pragma unroll
    for (int i = 0; i < 8; ++i) acc[i] = (v8f){};

    for (int kt = 0; kt < Kdim / 32; ++kt) {
        const int k0 = kt * 32;
        __builtin_prefetch(wrow0 + k0 + 64, 0, 1);
        v8us a0lo = *(const v8us*)(wrow0 + k0 + khalf * 2);
        v8us a0hi = *(const v8us*)(wrow0 + k0 + khalf * 2 + 8);
        v8us a1lo = *(const v8us*)(wrow1 + k0 + khalf * 2);
        v8us a1hi = *(const v8us*)(wrow1 + k0 + khalf * 2 + 8);
        v16us a0 = __builtin_shufflevector(a0lo, a0hi, 0,1,2,3,4,5,6,7,8,9,10,11,12,13,14,15);
        v16us a1 = __builtin_shufflevector(a1lo, a1hi, 0,1,2,3,4,5,6,7,8,9,10,11,12,13,14,15);
        const v16bf A0 = __builtin_bit_cast(v16bf, a0);
        const v16bf A1 = __builtin_bit_cast(v16bf, a1);
#pragma unroll
        for (int nt = 0; nt < 4; ++nt) {
            v16us bv = *(const v16us*)(bbase + nt * 16 * BROW + k0);
            const v16bf Bf = __builtin_bit_cast(v16bf, bv);
            acc[nt]     = __builtin_amdgcn_wmma_f32_16x16x32_bf16(
                              false, A0, false, Bf, (short)0, acc[nt], false, false);
            acc[4 + nt] = __builtin_amdgcn_wmma_f32_16x16x32_bf16(
                              false, A1, false, Bf, (short)0, acc[4 + nt], false, false);
        }
    }

    // ---- Stage 4: BN + SiLU epilogue, coalesced store ----
    // D layout (16x16 f32): VGPR r -> M = r (lanes 0-15) / 8+r (lanes 16-31),
    // N = lane & 15.
    const int col   = lane & 15;
    const int mside = (lane >> 4) * 8;

    float invs[2][8], biass[2][8];
#pragma unroll
    for (int r = 0; r < 8; ++r) {
        const int ch0 = m0 + mside + r;
        const int ch1 = ch0 + 16;
        invs[0][r]  = gamma[ch0] * rsqrtf(var[ch0] + 1e-5f);
        biass[0][r] = beta[ch0] - mean[ch0] * invs[0][r];
        invs[1][r]  = gamma[ch1] * rsqrtf(var[ch1] + 1e-5f);
        biass[1][r] = beta[ch1] - mean[ch1] * invs[1][r];
    }

#pragma unroll
    for (int nt = 0; nt < 4; ++nt) {
        const int w = w0 + nt * 16 + col;
#pragma unroll
        for (int r = 0; r < 8; ++r) {
            const int ch0 = m0 + mside + r;
            float o0 = acc[nt][r] * invs[0][r] + biass[0][r];
            o0 = o0 / (1.0f + __expf(-o0));
            out[(((size_t)b * C2n + ch0) * Hn + h) * Wn + w] = o0;

            float o1 = acc[4 + nt][r] * invs[1][r] + biass[1][r];
            o1 = o1 / (1.0f + __expf(-o1));
            out[(((size_t)b * C2n + ch0 + 16) * Hn + h) * Wn + w] = o1;
        }
    }
}

// ---------------------------------------------------------------------------
extern "C" void kernel_launch(void* const* d_in, const int* in_sizes, int n_in,
                              void* d_out, int out_size, void* d_ws, size_t ws_size,
                              hipStream_t stream) {
    const float* x      = (const float*)d_in[0];
    const float* w_off  = (const float*)d_in[1];
    const float* b_off  = (const float*)d_in[2];
    const float* w_main = (const float*)d_in[3];
    const float* gamma  = (const float*)d_in[4];
    const float* beta   = (const float*)d_in[5];
    const float* mean   = (const float*)d_in[6];
    const float* var    = (const float*)d_in[7];
    float* out = (float*)d_out;

    // Workspace layout: [offsets fp32 | bf16 main weights]
    float*          off_ws = (float*)d_ws;
    unsigned short* wbf    = (unsigned short*)((char*)d_ws +
                             (size_t)OFF_ELEMS * sizeof(float));

    akconv_convert_wmain<<<(WMAIN_ELEMS + 255) / 256, 256, 0, stream>>>(
        w_main, wbf, WMAIN_ELEMS);

    akconv_offset_conv<<<Bn * Hn, Wn, 0, stream>>>(x, w_off, b_off, off_ws);

    akconv_fused_gemm<<<(Bn * HWn) / POS, 256, 0, stream>>>(
        x, off_ws, wbf, gamma, beta, mean, var, out);
}
